// PolicyGNN_74552042324488
// MI455X (gfx1250) — matrix operand order repaired and checked by reading, not compiled, not run on hardware
//
#include <hip/hip_runtime.h>

typedef __attribute__((ext_vector_type(16))) _Float16 v16h;
typedef __attribute__((ext_vector_type(8)))  _Float16 v8h;
typedef __attribute__((ext_vector_type(8)))  float    v8f;

#define LN_EPS    1e-5f
#define INV_SQRT2 0.70710678118654752f

__device__ __forceinline__ float gelu_exact(float x) {
  return 0.5f * x * (1.0f + erff(x * INV_SQRT2));
}

// ---------------------------------------------------------------- prep kernels
__global__ void cvt_f16_kernel(const float* __restrict__ s, _Float16* __restrict__ d, int n) {
  int i = blockIdx.x * blockDim.x + threadIdx.x;
  if (i < n) d[i] = (_Float16)s[i];
}

// Repack row-major f32 B[K,Nc] into WMMA-B lane order (16x16x32 f16):
// lane half h holds K = 32*kt + 8*h + e + (e>=8 ? 8 : 0) (e=0..15), column n = lane%16.
// dst[((kt*2+h)*Nc + n)*16 + e] = B[(kt*32 + h*8 + e + (e>=8 ? 8 : 0))*Nc + n]
__global__ void repack_b_kernel(const float* __restrict__ src, _Float16* __restrict__ dst,
                                int K, int Nc) {
  int o = blockIdx.x * blockDim.x + threadIdx.x;
  if (o >= K * Nc) return;
  int e  = o & 15;
  int t  = o >> 4;
  int n  = t % Nc;
  int r  = t / Nc;
  int h  = r & 1;
  int kt = r >> 1;
  int k  = kt * 32 + h * 8 + e + (e >= 8 ? 8 : 0);
  dst[o] = (_Float16)src[k * Nc + n];
}

// ---------------------------------------------------------------- WMMA GEMM
// out = epi(A[M,K] @ B[K,Nc] + bias); A row-major f16, B packed (repack_b).
// One wave computes a 16x64 tile (4 WMMA accumulators reuse each A fragment).
// Requires Nc % 64 == 0, K % 32 == 0. epi: 0=none, 1=exact GELU.
__global__ __launch_bounds__(256) void gemm_wmma_kernel(
    const _Float16* __restrict__ A, const _Float16* __restrict__ Bp,
    const float* __restrict__ bias, float* __restrict__ outF32,
    _Float16* __restrict__ outF16, int M, int K, int Nc, int epi) {
  int ngN   = Nc >> 6;                          // 64-wide n-groups
  int tiles = (M >> 4) * ngN;
  int wid   = blockIdx.x * 8 + (threadIdx.x >> 5);
  if (wid >= tiles) return;
  int mt = wid / ngN, ng = wid - mt * ngN;
  int lane = threadIdx.x & 31;
  int h = lane >> 4, nlo = lane & 15, base0 = h << 3;
  int arow = mt * 16 + nlo;                     // A lane row: M = lane%16
  const _Float16* arp = A + (size_t)arow * K + base0;
  const _Float16* bp  = Bp + ((size_t)h * Nc + (size_t)ng * 64 + nlo) * 16;
  v8f Cz = {};
  v8f C[4] = {Cz, Cz, Cz, Cz};
  int kts = K >> 5;
  for (int kt = 0; kt < kts; ++kt) {
    // A lane K set: {base0..base0+7} U {16+base0..16+base0+7} -> two 16B loads
    v8h alo = *(const v8h*)(arp + kt * 32);
    v8h ahi = *(const v8h*)(arp + kt * 32 + 16);
    v16h a;
#pragma unroll
    for (int e = 0; e < 8; ++e) { a[e] = alo[e]; a[8 + e] = ahi[e]; }
    const _Float16* bkt = bp + (size_t)kt * 2 * Nc * 16;
#pragma unroll
    for (int t = 0; t < 4; ++t) {
      v16h b = *(const v16h*)(bkt + t * 16 * 16);
      C[t] = __builtin_amdgcn_wmma_f32_16x16x32_f16(false, a, false, b, (short)0, C[t],
                                                    false, false);
    }
  }
#pragma unroll
  for (int t = 0; t < 4; ++t) {
    int ncol = ng * 64 + t * 16 + nlo;          // C lane col: N = lane%16
    float bi = bias ? bias[ncol] : 0.0f;
#pragma unroll
    for (int p = 0; p < 8; ++p) {               // C VGPR p: M = p + 8h
      int m = mt * 16 + p + 8 * h;
      float x = C[t][p] + bi;
      if (epi == 1) x = gelu_exact(x);
      size_t oidx = (size_t)m * Nc + ncol;
      if (outF32) outF32[oidx] = x;
      if (outF16) outF16[oidx] = (_Float16)x;
    }
  }
}

// ---------------------------------------------------------------- LN / softmax
__device__ __forceinline__ float block_sum_256(float v, float* red) {
  int tid = threadIdx.x;
  red[tid] = v;
  __syncthreads();
  for (int s = 128; s > 0; s >>= 1) {
    if (tid < s) red[tid] += red[tid + s];
    __syncthreads();
  }
  float r = red[0];
  __syncthreads();
  return r;
}

__global__ __launch_bounds__(256) void ln256_f16_kernel(
    const float* __restrict__ in, const float* __restrict__ g,
    const float* __restrict__ be, _Float16* __restrict__ out16) {
  __shared__ float red[256];
  int row = blockIdx.x, tid = threadIdx.x;
  float x  = in[(size_t)row * 256 + tid];
  float s1 = block_sum_256(x, red);
  float s2 = block_sum_256(x * x, red);
  float mu   = s1 * (1.0f / 256.0f);
  float var  = s2 * (1.0f / 256.0f) - mu * mu;
  float rstd = rsqrtf(var + LN_EPS);
  out16[(size_t)row * 256 + tid] = (_Float16)((x - mu) * rstd * g[tid] + be[tid]);
}

__global__ __launch_bounds__(256) void softmax1024_kernel(float* __restrict__ a) {
  __shared__ float red[256];
  int row = blockIdx.x, tid = threadIdx.x;
  float* r = a + (size_t)row * 1024;
  float x0 = r[tid], x1 = r[tid + 256], x2 = r[tid + 512], x3 = r[tid + 768];
  float m = fmaxf(fmaxf(x0, x1), fmaxf(x2, x3));
  red[tid] = m; __syncthreads();
  for (int s = 128; s > 0; s >>= 1) {
    if (tid < s) red[tid] = fmaxf(red[tid], red[tid + s]);
    __syncthreads();
  }
  m = red[0]; __syncthreads();
  x0 = expf(x0 - m); x1 = expf(x1 - m); x2 = expf(x2 - m); x3 = expf(x3 - m);
  float inv = 1.0f / block_sum_256(x0 + x1 + x2 + x3, red);
  r[tid] = x0 * inv; r[tid + 256] = x1 * inv; r[tid + 512] = x2 * inv; r[tid + 768] = x3 * inv;
}

// ---------------------------------------------------------------- fused stage 3
// adj_out[i,:] = sum_j adj_wgt[i,j] * tanh( LN(gelu(u_i + v_j + b1)) @ W2 + b2 )
// One wave per i; 64 j-tiles of 16; h built directly in WMMA A layout.
// v-tiles are double-buffered in LDS via GLOBAL_LOAD_ASYNC_TO_LDS_B128 (ASYNCcnt),
// overlapping the global->LDS copy of tile jt+1 with the compute on tile jt.
__global__ __launch_bounds__(256) void stage3_fused_kernel(
    const float* __restrict__ u, const float* __restrict__ v,
    const float* __restrict__ wgt, const _Float16* __restrict__ W2qp,
    const float* __restrict__ b1q, const float* __restrict__ gq,
    const float* __restrict__ beq, const float* __restrict__ b2q,
    float* __restrict__ adj_out) {
  __shared__ float vtile[2][16 * 64];
  int tid  = threadIdx.x;
  int lane = tid & 31, w = tid >> 5;
  int i = blockIdx.x * 8 + w;
  int h = lane >> 4, nlo = lane & 15, base0 = h << 3;

  int   kmap[32];
  float uv[32], gv[32], bev[32];
#pragma unroll
  for (int c = 0; c < 2; ++c)
#pragma unroll
    for (int e = 0; e < 16; ++e) {
      int k = 32 * c + base0 + e + (e >= 8 ? 8 : 0);   // A-layout K map
      int q = c * 16 + e;
      kmap[q] = k;
      uv[q]  = u[i * 64 + k] + b1q[k];                 // fold b1 into u
      gv[q]  = gq[k];
      bev[q] = beq[k];
    }
  v16h B[4][2];                                        // W2q resident in regs
#pragma unroll
  for (int t = 0; t < 4; ++t)
#pragma unroll
    for (int c = 0; c < 2; ++c)
      B[t][c] = *(const v16h*)(W2qp + (((c * 2 + h) * 64) + t * 16 + nlo) * 16);
  float b2v[4];
#pragma unroll
  for (int t = 0; t < 4; ++t) b2v[t] = b2q[t * 16 + nlo];

  // async prologue: tile 0 -> buffer 0 (each thread pulls one 16B chunk)
  {
    unsigned lds0   = (unsigned)(size_t)&vtile[0][tid * 4];
    const float* g0 = v + tid * 4;
    asm volatile("global_load_async_to_lds_b128 %0, %1, off"
                 :: "v"(lds0), "v"(g0) : "memory");
  }

  float acc[4] = {0.f, 0.f, 0.f, 0.f};
  v8f Cz = {};
  for (int jt = 0; jt < 64; ++jt) {
    // prefetch tile jt+1 (wrap: last one is a harmless dummy, drained by endpgm)
    {
      int nb = (jt + 1) & 63;
      unsigned ldsn   = (unsigned)(size_t)&vtile[(jt + 1) & 1][tid * 4];
      const float* gn = v + nb * 1024 + tid * 4;
      asm volatile("global_load_async_to_lds_b128 %0, %1, off"
                   :: "v"(ldsn), "v"(gn) : "memory");
    }
    asm volatile("s_wait_asynccnt 0x1" ::: "memory");  // tile jt slices landed
    __syncthreads();                                   // ... in every wave
    const float* vt = vtile[jt & 1];

    // h row j = lane%16; lanes L and L^16 hold the two K-halves of a row
    float hbuf[32], s1 = 0.f, s2 = 0.f;
#pragma unroll
    for (int q = 0; q < 32; ++q) {
      float x = gelu_exact(uv[q] + vt[nlo * 64 + kmap[q]]);
      hbuf[q] = x; s1 += x; s2 += x * x;
    }
    s1 += __shfl_xor(s1, 16);
    s2 += __shfl_xor(s2, 16);
    float mu   = s1 * (1.0f / 64.0f);
    float var  = s2 * (1.0f / 64.0f) - mu * mu;
    float rstd = rsqrtf(var + LN_EPS);
    v16h a0, a1;
#pragma unroll
    for (int e = 0; e < 16; ++e) {
      a0[e] = (_Float16)((hbuf[e]      - mu) * rstd * gv[e]      + bev[e]);
      a1[e] = (_Float16)((hbuf[16 + e] - mu) * rstd * gv[16 + e] + bev[16 + e]);
    }
    v8f C[4] = {Cz, Cz, Cz, Cz};
#pragma unroll
    for (int t = 0; t < 4; ++t) {
      C[t] = __builtin_amdgcn_wmma_f32_16x16x32_f16(false, a0, false, B[t][0], (short)0,
                                                    C[t], false, false);
      C[t] = __builtin_amdgcn_wmma_f32_16x16x32_f16(false, a1, false, B[t][1], (short)0,
                                                    C[t], false, false);
    }
    float wv[8];
#pragma unroll
    for (int p = 0; p < 8; ++p) wv[p] = wgt[(size_t)i * 1024 + jt * 16 + 8 * h + p];
#pragma unroll
    for (int t = 0; t < 4; ++t) {
      float part = 0.f;
#pragma unroll
      for (int p = 0; p < 8; ++p) part += wv[p] * tanhf(C[t][p] + b2v[t]);
      part += __shfl_xor(part, 16);   // other half's 8 rows (same column)
      acc[t] += part;
    }
    __syncthreads();                  // all reads of vtile[jt&1] done before overwrite
  }
  if (h == 0) {
#pragma unroll
    for (int t = 0; t < 4; ++t) adj_out[i * 64 + t * 16 + nlo] = acc[t];
  }
}

// ---------------------------------------------------------------- stage 4: scan
__global__ __launch_bounds__(256) void stage4_scan_kernel(
    const float* __restrict__ xs_in, const int* __restrict__ ids,
    const float* __restrict__ adj_out, const float* __restrict__ W1,
    const float* __restrict__ b1, const float* __restrict__ g,
    const float* __restrict__ be, const float* __restrict__ W2,
    const float* __restrict__ b2, float* __restrict__ xs_out) {
  __shared__ float xsl[1024];
  __shared__ float red[256];
  int tid = threadIdx.x;
  for (int p = tid; p < 1024; p += 256) xsl[p] = xs_in[p];
  __syncthreads();
  float b2s = b2[0];
  for (int t = 0; t < 1024; ++t) {
    int idx = ids[t];
    float acc = b1[tid];                       // thread tid owns hidden unit tid
    for (int k = 0; k < 1024; ++k) acc += xsl[k] * W1[k * 256 + tid];
    const float* ao  = adj_out + (size_t)idx * 64;
    const float* W1t = W1 + 1024 * 256;
    for (int k = 0; k < 64; ++k) acc += ao[k] * W1t[k * 256 + tid];
    float hx  = gelu_exact(acc);
    float s1  = block_sum_256(hx, red);
    float s2  = block_sum_256(hx * hx, red);
    float mu  = s1 * (1.0f / 256.0f);
    float var = s2 * (1.0f / 256.0f) - mu * mu;
    float hn  = (hx - mu) * rsqrtf(var + LN_EPS) * g[tid] + be[tid];
    float s   = block_sum_256(hn * W2[tid], red);
    if (tid == 0) xsl[idx] = 1.0f / (1.0f + expf(-(s + b2s)));
    __syncthreads();
  }
  for (int p = tid; p < 1024; p += 256) xs_out[p] = xsl[p];
}

// ---------------------------------------------------------------- launcher
extern "C" void kernel_launch(void* const* d_in, const int* in_sizes, int n_in,
                              void* d_out, int out_size, void* d_ws, size_t ws_size,
                              hipStream_t stream) {
  (void)in_sizes; (void)n_in; (void)out_size; (void)ws_size;
  const float* adj = (const float*)d_in[0];
  const float* xs  = (const float*)d_in[1];
  const int*   ids = (const int*)d_in[2];
  const float* AW1 = (const float*)d_in[3];  const float* Ab1 = (const float*)d_in[4];
  const float* Ag  = (const float*)d_in[5];  const float* Abe = (const float*)d_in[6];
  const float* AW2 = (const float*)d_in[7];  const float* Ab2 = (const float*)d_in[8];
  const float* BW1 = (const float*)d_in[9];  const float* Bb1 = (const float*)d_in[10];
  const float* Bg  = (const float*)d_in[11]; const float* Bbe = (const float*)d_in[12];
  const float* BW2 = (const float*)d_in[13]; const float* Bb2 = (const float*)d_in[14];
  const float* QW1 = (const float*)d_in[15]; const float* Qb1 = (const float*)d_in[16];
  const float* Qg  = (const float*)d_in[17]; const float* Qbe = (const float*)d_in[18];
  const float* QW2 = (const float*)d_in[19]; const float* Qb2 = (const float*)d_in[20];
  const float* NW1 = (const float*)d_in[21]; const float* Nb1 = (const float*)d_in[22];
  const float* Ng  = (const float*)d_in[23]; const float* Nbe = (const float*)d_in[24];
  const float* NW2 = (const float*)d_in[25]; const float* Nb2 = (const float*)d_in[26];

  char* ws = (char*)d_ws;
  size_t off = 0;
  auto alloc = [&](size_t bytes) -> char* {
    char* p = ws + off;
    off += (bytes + 255) & ~(size_t)255;
    return p;
  };
  _Float16* adjH    = (_Float16*)alloc(1024 * 1024 * 2);
  _Float16* W1pA    = (_Float16*)alloc(1024 * 256 * 2);
  _Float16* W2pA    = (_Float16*)alloc(256 * 64 * 2);
  _Float16* W1pB    = (_Float16*)alloc(64 * 256 * 2);
  _Float16* W2pB    = (_Float16*)alloc(256 * 1024 * 2);
  _Float16* W1pQa   = (_Float16*)alloc(64 * 64 * 2);
  _Float16* W1pQb   = (_Float16*)alloc(64 * 64 * 2);
  _Float16* W2pQ    = (_Float16*)alloc(64 * 64 * 2);
  float*    h1      = (float*)alloc(1024 * 256 * 4);
  _Float16* h1ln    = (_Float16*)alloc(1024 * 256 * 2);
  float*    adjInp  = (float*)alloc(1024 * 64 * 4);
  _Float16* adjInpH = (_Float16*)alloc(1024 * 64 * 2);
  float*    h2      = (float*)alloc(1024 * 256 * 4);
  _Float16* h2ln    = (_Float16*)alloc(1024 * 256 * 2);
  float*    wgt     = (float*)alloc(1024 * 1024 * 4);   // logits -> softmax in-place
  float*    uBuf    = (float*)alloc(1024 * 64 * 4);
  float*    vBuf    = (float*)alloc(1024 * 64 * 4);
  float*    adjOut  = (float*)alloc(1024 * 64 * 4);

  // prep: f16 convert + B repacks
  cvt_f16_kernel<<<4096, 256, 0, stream>>>(adj, adjH, 1024 * 1024);
  repack_b_kernel<<<(1024 * 256 + 255) / 256, 256, 0, stream>>>(AW1, W1pA, 1024, 256);
  repack_b_kernel<<<(256 * 64 + 255) / 256, 256, 0, stream>>>(AW2, W2pA, 256, 64);
  repack_b_kernel<<<(64 * 256 + 255) / 256, 256, 0, stream>>>(BW1, W1pB, 64, 256);
  repack_b_kernel<<<(256 * 1024 + 255) / 256, 256, 0, stream>>>(BW2, W2pB, 256, 1024);
  repack_b_kernel<<<(64 * 64 + 255) / 256, 256, 0, stream>>>(QW1, W1pQa, 64, 64);
  repack_b_kernel<<<(64 * 64 + 255) / 256, 256, 0, stream>>>(QW1 + 64 * 64, W1pQb, 64, 64);
  repack_b_kernel<<<(64 * 64 + 255) / 256, 256, 0, stream>>>(QW2, W2pQ, 64, 64);

  // stage 1: adj_inp = gelu(LN(gelu(adj@W1+b1))@W2+b2)   [tiles = (M/16)*(Nc/64)]
  gemm_wmma_kernel<<<32, 256, 0, stream>>>(adjH, W1pA, Ab1, h1, nullptr, 1024, 1024, 256, 1);
  ln256_f16_kernel<<<1024, 256, 0, stream>>>(h1, Ag, Abe, h1ln);
  gemm_wmma_kernel<<<8, 256, 0, stream>>>(h1ln, W2pA, Ab2, adjInp, adjInpH, 1024, 256, 64, 1);

  // stage 2: adj_wgt = softmax(LN(gelu(adj_inp@W1+b1))@W2+b2)
  gemm_wmma_kernel<<<32, 256, 0, stream>>>(adjInpH, W1pB, Bb1, h2, nullptr, 1024, 64, 256, 1);
  ln256_f16_kernel<<<1024, 256, 0, stream>>>(h2, Bg, Bbe, h2ln);
  gemm_wmma_kernel<<<128, 256, 0, stream>>>(h2ln, W2pB, Bb2, wgt, nullptr, 1024, 256, 1024, 0);
  softmax1024_kernel<<<1024, 256, 0, stream>>>(wgt);

  // stage 3: u/v projections + fused pairwise query
  gemm_wmma_kernel<<<8, 256, 0, stream>>>(adjInpH, W1pQa, nullptr, uBuf, nullptr, 1024, 64, 64, 0);
  gemm_wmma_kernel<<<8, 256, 0, stream>>>(adjInpH, W1pQb, nullptr, vBuf, nullptr, 1024, 64, 64, 0);
  stage3_fused_kernel<<<128, 256, 0, stream>>>(uBuf, vBuf, wgt, W2pQ, Qb1, Qg, Qbe, Qb2, adjOut);

  // stage 4: sequential autoregressive scan
  stage4_scan_kernel<<<1, 256, 0, stream>>>(xs, ids, adjOut, NW1, Nb1, Ng, Nbe, NW2, Nb2,
                                            (float*)d_out);
}
